// DGPLoss_34505767256557
// MI455X (gfx1250) — compile-verified, matrix operand order checked by reading, not run on hardware
//
#include <hip/hip_runtime.h>
#include <hip/hip_bf16.h>

typedef __attribute__((ext_vector_type(16))) __bf16        v16bf;
typedef __attribute__((ext_vector_type(8)))  float         v8f;
typedef __attribute__((ext_vector_type(8)))  unsigned int  v8u;

#define HS 128
#define WS 256
#define HD 512
#define WD 1024
#define NB 2
#define NC 32
#define NOFF 24

// ---------------- kernel 0: zero the 4 accumulators ----------------
__global__ void k_init(float* acc) {
    if (threadIdx.x < 4) acc[threadIdx.x] = 0.0f;
}

// ---------------- kernel 1: 4x4 block medians of normalized disparity ----------------
// med[b][i][j] = lower-median of the >EPS normalized disparities in depth block (i,j), else 0
__global__ void k_blockmed(const float* __restrict__ dep, float* __restrict__ med, int total) {
    int t = blockIdx.x * blockDim.x + threadIdx.x;
    if (t >= total) return;                 // total = NB*HS*WS (one thread per 4x4 block)
    int b = t >> 15;                        // / (HS*WS)
    int r = t & (HS * WS - 1);
    int i = r >> 8;                         // / WS
    int j = r & (WS - 1);

    const float* base = dep + ((size_t)b * HD + (size_t)i * 4) * WD + (size_t)j * 4;
    float v[16];
#pragma unroll
    for (int rr = 0; rr < 4; ++rr) {
        float4 q = *reinterpret_cast<const float4*>(base + (size_t)rr * WD);
        v[rr * 4 + 0] = q.x; v[rr * 4 + 1] = q.y; v[rr * 4 + 2] = q.z; v[rr * 4 + 3] = q.w;
    }

    const float INF = __builtin_inff();
    int k = 0;
#pragma unroll
    for (int e = 0; e < 16; ++e) {
        float d = v[e];
        bool validd = (d >= 1.0f) && (d <= 100.0f);
        // normalized disparity: (1/d - 1/100) / (1/1 - 1/100); invalid depth -> 0
        float n = validd ? (1.0f / d - 0.01f) * (1.0f / 0.99f) : 0.0f;
        bool va = n > 1e-8f;                // matches (dpatch > EPS_FN)
        k += va ? 1 : 0;
        v[e] = va ? n : INF;
    }
    int m = (k > 0) ? ((k - 1) >> 1) : 0;

    // branch-free rank selection (ties broken by index -> unique ranks, register-only)
    float medv = 0.0f;
#pragma unroll
    for (int a = 0; a < 16; ++a) {
        int rank = 0;
#pragma unroll
        for (int bb = 0; bb < 16; ++bb) {
            bool less = (v[bb] < v[a]) || ((v[bb] == v[a]) && (bb < a));
            rank += less ? 1 : 0;
        }
        medv = (rank == m) ? v[a] : medv;
    }
    med[t] = (k > 0) ? medv : 0.0f;         // patch_ok == (k >= 1)
}

// ---------------- kernel 2: per-pixel L2-normalize features, pack bf16 (B,H,W,C) ----------------
__device__ __forceinline__ unsigned int f2bf_rne(float f) {
    unsigned int u = __float_as_uint(f);
    return (u + 0x7FFFu + ((u >> 16) & 1u)) >> 16;
}

__global__ void k_norm(const float* __restrict__ seg, unsigned short* __restrict__ sfb, int total) {
    int t = blockIdx.x * blockDim.x + threadIdx.x;
    if (t >= total) return;                 // total = NB*HS*WS (one thread per pixel)
    int b = t >> 15;
    int r = t & (HS * WS - 1);
    const float* p = seg + (size_t)b * NC * HS * WS + r;

    float x[NC];
    float n2 = 0.0f;
#pragma unroll
    for (int c = 0; c < NC; ++c) { x[c] = p[(size_t)c * HS * WS]; n2 += x[c] * x[c]; }
    float inv = 1.0f / fmaxf(sqrtf(n2), 1e-12f);

    unsigned int w[NC / 2];
#pragma unroll
    for (int c = 0; c < NC; c += 2) {
        unsigned int lo = f2bf_rne(x[c] * inv);
        unsigned int hi = f2bf_rne(x[c + 1] * inv);
        w[c >> 1] = lo | (hi << 16);
    }
    uint4* q = reinterpret_cast<uint4*>(sfb + (size_t)t * NC);
#pragma unroll
    for (int c = 0; c < 4; ++c)
        q[c] = make_uint4(w[4 * c + 0], w[4 * c + 1], w[4 * c + 2], w[4 * c + 3]);
}

// ---------------- kernel 3: WMMA neighbor-dot kernel (branchless, pipelined) ----------------
// Per-lane A/B fragment load for v_wmma_f32_16x16x32_bf16:
// lanes 0-15 hold K[0..8)+K[16..24), lanes 16-31 hold K[8..16)+K[24..32) of row M=lane%16.
__device__ __forceinline__ v16bf load_frag(const unsigned short* row, int k0) {
    uint4 a = *reinterpret_cast<const uint4*>(row + k0);
    uint4 c = *reinterpret_cast<const uint4*>(row + k0 + 16);
    v8u u;
    u[0] = a.x; u[1] = a.y; u[2] = a.z; u[3] = a.w;
    u[4] = c.x; u[5] = c.y; u[6] = c.z; u[7] = c.w;
    return __builtin_bit_cast(v16bf, u);
}

__device__ __forceinline__ float vsel8(v8f d, int c) {
    float r = d[0];
    r = (c == 1) ? d[1] : r; r = (c == 2) ? d[2] : r; r = (c == 3) ? d[3] : r;
    r = (c == 4) ? d[4] : r; r = (c == 5) ? d[5] : r; r = (c == 6) ? d[6] : r;
    r = (c == 7) ? d[7] : r;
    return r;
}

// offset index o in [0,24) -> (di,dj), skipping the center cell (compile-time)
#define ODI(o) ((((o) < 12) ? (o) : (o) + 1) / 5 - 2)
#define ODJ(o) ((((o) < 12) ? (o) : (o) + 1) % 5 - 2)

__global__ void k_main(const unsigned short* __restrict__ sfb,
                       const float* __restrict__ med,
                       float* __restrict__ acc, int nwaves) {
    int wid  = (blockIdx.x * blockDim.x + threadIdx.x) >> 5;
    int lane = threadIdx.x & 31;
    if (wid >= nwaves) return;              // wave-uniform: EXEC stays all-ones for WMMA

    int tile = wid & 15;                    // 16 tiles of 16 anchors cover j in [2,254)
    int rest = wid >> 4;
    int i = (rest % 124) + 2;               // anchor row in [2,126)
    int b = rest / 124;
    int j0 = 2 + tile * 16;

    int m  = lane & 15;                     // pixel slot this lane loads for fragments
    int jA = j0 + m; jA = (jA > 253) ? 253 : jA;   // clamp (last tile partially masked)
    int k0 = (lane < 16) ? 0 : 8;

    const unsigned short* sfb_b = sfb + (size_t)b * HS * WS * NC;
    const float*          med_b = med + (size_t)b * HS * WS;

    __builtin_prefetch(sfb_b + ((size_t)i * WS + j0) * NC, 0, 1);   // global_prefetch_b8

    // diagonal ownership of the 16x16 f32 D fragment (branchless mask)
    bool own  = (lane < 8) || (lane >= 24);
    int  dm   = (lane < 8) ? lane : ((lane >= 24) ? (lane - 16) : 0);  // anchor index in tile
    int  cidx = (lane < 8) ? lane : ((lane >= 24) ? (lane - 24) : 0);  // vgpr slot of D[dm][dm]
    int  jm   = j0 + dm;
    float wmask = (own && (jm <= 253)) ? 1.0f : 0.0f;
    int  jmc  = (jm > 253) ? 253 : jm;

    float d_c = med_b[(size_t)i * WS + jmc];

    // ---- preload all 24 neighbor medians (independent loads, issue as one clause) ----
    float medn[NOFF];
#pragma unroll
    for (int o = 0; o < NOFF; ++o) {
        const int di = ODI(o), dj = ODJ(o);
        int jn = jmc + dj; jn = (jn < 0) ? 0 : ((jn > 255) ? 255 : jn);
        medn[o] = med_b[(size_t)(i + di) * WS + jn];
    }

    // ---- A fragment (center tile), and first B fragment ----
    v16bf A  = load_frag(sfb_b + ((size_t)i * WS + jA) * NC, k0);

    int jB0 = j0 + ODJ(0) + m; jB0 = (jB0 < 0) ? 0 : ((jB0 > 255) ? 255 : jB0);
    v16bf Bf = load_frag(sfb_b + ((size_t)(i + ODI(0)) * WS + jB0) * NC, k0);

    float lsum = 0.0f, vcnt = 0.0f;

    // ---- software-pipelined WMMA loop: load B(o+1) while WMMA(o) executes ----
#pragma unroll
    for (int o = 0; o < NOFF; ++o) {
        v16bf Bnext = Bf;
        if (o + 1 < NOFF) {
            const int din = ODI(o + 1), djn = ODJ(o + 1);
            int jBn = j0 + djn + m; jBn = (jBn < 0) ? 0 : ((jBn > 255) ? 255 : jBn);
            Bnext = load_frag(sfb_b + ((size_t)(i + din) * WS + jBn) * NC, k0);
        }

        v8f C = {};
        v8f D = __builtin_amdgcn_wmma_f32_16x16x32_bf16(
            /*neg_a=*/false, A, /*neg_b=*/false, Bf,
            /*c_mod=*/(short)0, C, /*reuse_a=*/false, /*reuse_b=*/false);

        // branchless scalar tail (all lanes; masked accumulate)
        float dot = vsel8(D, cidx);                     // <s_center, s_neighbor>
        float d_n = medn[o];
        float sd2 = fmaxf(2.0f - 2.0f * dot, 0.0f);     // ||s_c - s_n||^2 for unit vectors
        // exp(-|dc-dn|/TAU) * exp(-sd2) fused into a single transcendental
        lsum += wmask * __expf(-(fabsf(d_c - d_n) + sd2));
        vcnt += wmask * (((d_c > 0.0f) && (d_n > 0.0f)) ? 1.0f : 0.0f);

        Bf = Bnext;
    }

    // wave32 reduction, then one pair of float atomics per wave (b is wave-uniform)
#pragma unroll
    for (int off = 16; off > 0; off >>= 1) {
        lsum += __shfl_down(lsum, off, 32);
        vcnt += __shfl_down(vcnt, off, 32);
    }
    if (lane == 0) {
        atomicAdd(&acc[b], lsum);
        atomicAdd(&acc[2 + b], vcnt);
    }
}

// ---------------- kernel 4: finalize scalar ----------------
__global__ void k_final(const float* __restrict__ acc, float* __restrict__ out) {
    if (threadIdx.x == 0) {
        float p0 = acc[0] / fmaxf(acc[2], 1.0f);
        float p1 = acc[1] / fmaxf(acc[3], 1.0f);
        out[0] = 0.5f * (p0 + p1);          // mean over B, SCALE = 1
    }
}

extern "C" void kernel_launch(void* const* d_in, const int* in_sizes, int n_in,
                              void* d_out, int out_size, void* d_ws, size_t ws_size,
                              hipStream_t stream) {
    const float* seg = (const float*)d_in[0];   // (2,32,128,256) f32
    const float* dep = (const float*)d_in[1];   // (2,512,1024)  f32
    float* out = (float*)d_out;                 // scalar f32

    char* ws = (char*)d_ws;
    float*          acc = (float*)ws;                              // 16 B
    float*          med = (float*)(ws + 256);                      // 2*128*256*4 = 256 KB
    unsigned short* sfb = (unsigned short*)(ws + 256 + NB * HS * WS * 4); // 4 MB

    const int npix = NB * HS * WS;          // 65536
    k_init    <<<1,   32,  0, stream>>>(acc);
    k_blockmed<<<npix / 256, 256, 0, stream>>>(dep, med, npix);
    k_norm    <<<npix / 256, 256, 0, stream>>>(seg, sfb, npix);

    const int nwaves = NB * 124 * 16;       // 3968 waves, 8 waves/block
    k_main    <<<nwaves / 8, 256, 0, stream>>>(sfb, med, acc, nwaves);
    k_final   <<<1,   32,  0, stream>>>(acc, out);
}
